// snn_layer_30270929502964
// MI455X (gfx1250) — matrix operand order repaired and checked by reading, not compiled, not run on hardware
//
#include <hip/hip_runtime.h>

typedef __attribute__((ext_vector_type(16))) __bf16 v16bf;
typedef __attribute__((ext_vector_type(4)))  __bf16 v4bf;
typedef __attribute__((ext_vector_type(8)))  float  v8f;
typedef __attribute__((ext_vector_type(4)))  float  f32x4v;   // true ext-vector for NT stores

constexpr int BATCH = 128;
constexpr int FEAT  = 512;   // K
constexpr int UNITS = 1024;  // N
constexpr int TIME  = 256;   // M
constexpr int BM = 128, BN = 128, BK = 32;
constexpr int LD = BK + 8;   // padded LDS row in bf16 elems (80 B; keeps b64/b128 alignment)

__device__ __forceinline__ float f4_at(const float4& v, int i) {
    return i == 0 ? v.x : i == 1 ? v.y : i == 2 ? v.z : v.w;
}

// ---- CDNA5 async copy: 16B global -> LDS, tracked by ASYNCcnt ----
__device__ __forceinline__ void async_cp16(const __bf16* g, __bf16* l) {
    unsigned           la = (unsigned)(size_t)(const void*)l;       // LDS byte address
    unsigned long long ga = (unsigned long long)(size_t)(const void*)g;
    asm volatile("global_load_async_to_lds_b128 %0, %1, off"
                 :: "v"(la), "v"(ga) : "memory");
}
__device__ __forceinline__ void wait_async0() {
    asm volatile("s_wait_asynccnt 0x0" ::: "memory");
}

// ---- shared fragment-load + WMMA body (ISA §7.12.2 layouts) ----
__device__ __forceinline__ void wmma_tile(const __bf16* A, const __bf16* B,
                                          int wm, int wn, int half, int lr,
                                          v8f acc[4][2]) {
    v16bf af[4], bfv[2];
#pragma unroll
    for (int i = 0; i < 4; ++i) {
        // A 16x32 bf16: lanes 0-15 K={0..7,16..23}, lanes 16-31 K={8..15,24..31}
        const __bf16* row = A + (wm * 64 + i * 16 + lr) * LD + half * 8;
        union { v16bf v; float4 q[2]; } u;
        u.q[0] = *reinterpret_cast<const float4*>(row);
        u.q[1] = *reinterpret_cast<const float4*>(row + 16);
        af[i] = u.v;
    }
#pragma unroll
    for (int j = 0; j < 2; ++j) {
        // B 32x16 bf16: lanes 0-15 K={0..7,16..23}, lanes 16-31 K={8..15,24..31}
        const __bf16* row = B + (wn * 32 + j * 16 + lr) * LD + half * 8;
        union { v16bf v; float4 q[2]; } u;
        u.q[0] = *reinterpret_cast<const float4*>(row);
        u.q[1] = *reinterpret_cast<const float4*>(row + 16);
        bfv[j] = u.v;
    }
#pragma unroll
    for (int i = 0; i < 4; ++i)
#pragma unroll
        for (int j = 0; j < 2; ++j)
            acc[i][j] = __builtin_amdgcn_wmma_f32_16x16x32_bf16(
                false, af[i], false, bfv[j], (short)0, acc[i][j], false, false);
}

// Output is a 134MB write-once stream: store non-temporal so it doesn't evict
// the bf16 operand working set (fits in 192MB L2) that the GEMM re-reads.
__device__ __forceinline__ void epilogue(float* __restrict__ out, int b, int m0, int u0,
                                         int wm, int wn, int half, int lr,
                                         v8f acc[4][2]) {
#pragma unroll
    for (int i = 0; i < 4; ++i) {
        const int t0 = m0 + wm * 64 + i * 16 + half * 8;   // C VGPR r -> M = r + 8*half
#pragma unroll
        for (int j = 0; j < 2; ++j) {
            const int n = u0 + wn * 32 + j * 16 + lr;      // C lane -> N = lr
            float r[8];
#pragma unroll
            for (int e = 0; e < 8; ++e) r[e] = (acc[i][j][e] > 1.0f) ? 1.0f : 0.0f;
            f32x4v lo = { r[0], r[1], r[2], r[3] };
            f32x4v hi = { r[4], r[5], r[6], r[7] };
            f32x4v* p = reinterpret_cast<f32x4v*>(out + ((size_t)b * UNITS + n) * TIME + t0);
            __builtin_nontemporal_store(lo, p);
            __builtin_nontemporal_store(hi, p + 1);
        }
    }
}

// =====================================================================
// Phase 1: transpose + f32->bf16 convert.  in[R][C] f32 -> out[C][R] bf16
// (batched over blockIdx.z with stride R*C). Raw f32 input is read-once
// and dead afterwards -> non-temporal loads.
// =====================================================================
__global__ __launch_bounds__(256)
void transpose_cvt_kernel(const float* __restrict__ in, __bf16* __restrict__ outp,
                          int R, int C) {
    __shared__ float tile[32][33];
    const size_t base = (size_t)blockIdx.z * R * C;
    const int r0 = blockIdx.y * 32, c0 = blockIdx.x * 32;
    const int tx = threadIdx.x, ty = threadIdx.y;      // (32, 8)
#pragma unroll
    for (int i = 0; i < 32; i += 8)
        tile[ty + i][tx] =
            __builtin_nontemporal_load(&in[base + (size_t)(r0 + ty + i) * C + c0 + tx]);
    __syncthreads();
#pragma unroll
    for (int i = 0; i < 32; i += 8)
        outp[base + (size_t)(c0 + ty + i) * R + r0 + tx] = (__bf16)tile[tx][ty + i];
}

// =====================================================================
// Phase 2: GEMM with ASYNCcnt-pipelined global->LDS staging (bf16 operands)
//   xt [B, T, F] bf16 (K contiguous), wt [U, F] bf16 (K contiguous)
// =====================================================================
__global__ __launch_bounds__(256)
void snn_wmma_async_kernel(const __bf16* __restrict__ xt,
                           const __bf16* __restrict__ wt,
                           float* __restrict__ out) {
    __shared__ __align__(16) __bf16 As[2][BM * LD];
    __shared__ __align__(16) __bf16 Bs[2][BN * LD];

    const int u0 = blockIdx.x * BN;
    const int m0 = blockIdx.y * BM;
    const int b  = blockIdx.z;

    const int tid  = threadIdx.x;
    const int lane = tid & 31, wv = tid >> 5;
    const int wm = wv >> 2, wn = wv & 3;
    const int half = lane >> 4, lr = lane & 15;

    const int sr = tid >> 1;   // 0..127: tile row (t or u)
    const int sh = tid & 1;    // 0..1  : 32B half of the 64B row

    v8f acc[4][2];
#pragma unroll
    for (int i = 0; i < 4; ++i)
#pragma unroll
        for (int j = 0; j < 2; ++j)
#pragma unroll
            for (int e = 0; e < 8; ++e) acc[i][j][e] = 0.0f;

    const __bf16* ga0 = xt + ((size_t)b * TIME + m0 + sr) * FEAT + sh * 16;
    const __bf16* gb0 = wt + (size_t)(u0 + sr) * FEAT + sh * 16;

    auto stage = [&](__bf16* A, __bf16* B, int k0) {
        const __bf16* ga = ga0 + k0;
        const __bf16* gb = gb0 + k0;
        __bf16* la = A + sr * LD + sh * 16;
        __bf16* lb = B + sr * LD + sh * 16;
        async_cp16(ga,     la);
        async_cp16(ga + 8, la + 8);
        async_cp16(gb,     lb);
        async_cp16(gb + 8, lb + 8);
    };

    // pipeline: wait(tile n) -> barrier -> issue asyncs(tile n+1) -> compute(tile n)
    stage(As[0], Bs[0], 0);
    for (int k0 = 0; k0 < FEAT; k0 += 2 * BK) {
        wait_async0();
        __syncthreads();
        if (k0 + BK < FEAT) stage(As[1], Bs[1], k0 + BK);
        wmma_tile(As[0], Bs[0], wm, wn, half, lr, acc);

        wait_async0();
        __syncthreads();
        if (k0 + 2 * BK < FEAT) stage(As[0], Bs[0], k0 + 2 * BK);
        wmma_tile(As[1], Bs[1], wm, wn, half, lr, acc);
    }

    epilogue(out, b, m0, u0, wm, wn, half, lr, acc);
}

// =====================================================================
// Fallback: register-convert staging, no workspace needed
// =====================================================================
__global__ __launch_bounds__(256)
void snn_wmma_kernel(const float* __restrict__ x, const float* __restrict__ w,
                     float* __restrict__ out) {
    __shared__ __align__(16) __bf16 As[2][BM * LD];
    __shared__ __align__(16) __bf16 Bs[2][BN * LD];

    const int u0 = blockIdx.x * BN;
    const int m0 = blockIdx.y * BM;
    const int b  = blockIdx.z;

    const int tid  = threadIdx.x;
    const int lane = tid & 31, wv = tid >> 5;
    const int wm = wv >> 2, wn = wv & 3;
    const int half = lane >> 4, lr = lane & 15;

    const int kg = tid >> 5;   // 0..7  -> k = kg*4..+3
    const int tg = tid & 31;   // 0..31 -> 4 rows of t / u

    v8f acc[4][2];
#pragma unroll
    for (int i = 0; i < 4; ++i)
#pragma unroll
        for (int j = 0; j < 2; ++j)
#pragma unroll
            for (int e = 0; e < 8; ++e) acc[i][j][e] = 0.0f;

    const float* xg = x + (size_t)b * FEAT * TIME + m0 + tg * 4;
    const float* wg = w + u0 + tg * 4;

    float4 ar[4], br[4];
    auto fetch = [&](int k0) {
#pragma unroll
        for (int i = 0; i < 4; ++i)
            ar[i] = *reinterpret_cast<const float4*>(xg + (size_t)(k0 + kg * 4 + i) * TIME);
#pragma unroll
        for (int i = 0; i < 4; ++i)
            br[i] = *reinterpret_cast<const float4*>(wg + (size_t)(k0 + kg * 4 + i) * UNITS);
    };
    auto stage = [&](__bf16* A, __bf16* B) {
#pragma unroll
        for (int tt = 0; tt < 4; ++tt) {
            v4bf pa, pb;
#pragma unroll
            for (int i = 0; i < 4; ++i) {
                pa[i] = (__bf16)f4_at(ar[i], tt);
                pb[i] = (__bf16)f4_at(br[i], tt);
            }
            *reinterpret_cast<v4bf*>(A + (tg * 4 + tt) * LD + kg * 4) = pa;
            *reinterpret_cast<v4bf*>(B + (tg * 4 + tt) * LD + kg * 4) = pb;
        }
    };

    fetch(0);
    for (int k0 = 0; k0 < FEAT; k0 += 2 * BK) {
        stage(As[0], Bs[0]);
        fetch(k0 + BK);
        __syncthreads();
        wmma_tile(As[0], Bs[0], wm, wn, half, lr, acc);

        stage(As[1], Bs[1]);
        if (k0 + 2 * BK < FEAT) fetch(k0 + 2 * BK);
        __syncthreads();
        wmma_tile(As[1], Bs[1], wm, wn, half, lr, acc);
    }

    epilogue(out, b, m0, u0, wm, wn, half, lr, acc);
}

extern "C" void kernel_launch(void* const* d_in, const int* in_sizes, int n_in,
                              void* d_out, int out_size, void* d_ws, size_t ws_size,
                              hipStream_t stream) {
    const float* x = (const float*)d_in[0];   // inputs [B, F, T]
    const float* w = (const float*)d_in[1];   // w      [F, U]
    float* o       = (float*)d_out;           // out    [B, U, T]

    const size_t xt_bytes = (size_t)BATCH * TIME * FEAT * sizeof(__bf16);
    const size_t wt_bytes = (size_t)UNITS * FEAT * sizeof(__bf16);

    dim3 grid(UNITS / BN, TIME / BM, BATCH);  // (8, 2, 128)
    dim3 block(256);

    if (ws_size >= xt_bytes + wt_bytes) {
        __bf16* xt = (__bf16*)d_ws;                              // [B, T, F]
        __bf16* wt = (__bf16*)((char*)d_ws + xt_bytes);          // [U, F]

        // Phase 1: transpose-convert both operands to K-contiguous bf16
        dim3 tb(32, 8);
        dim3 tgx(TIME / 32, FEAT / 32, BATCH);                   // x: [F,T] -> [T,F]
        hipLaunchKernelGGL(transpose_cvt_kernel, tgx, tb, 0, stream, x, xt, FEAT, TIME);
        dim3 tgw(UNITS / 32, FEAT / 32, 1);                      // w: [F,U] -> [U,F]
        hipLaunchKernelGGL(transpose_cvt_kernel, tgw, tb, 0, stream, w, wt, FEAT, UNITS);

        // Phase 2: async-staged WMMA GEMM
        hipLaunchKernelGGL(snn_wmma_async_kernel, grid, block, 0, stream, xt, wt, o);
    } else {
        hipLaunchKernelGGL(snn_wmma_kernel, grid, block, 0, stream, x, w, o);
    }
}
